// Update_12584254177896
// MI455X (gfx1250) — compile-verified, matrix operand order checked by reading, not run on hardware
//
#include <hip/hip_runtime.h>
#include <hip/hip_fp16.h>

typedef _Float16 half_t;
typedef __attribute__((ext_vector_type(16))) _Float16 v16h;
typedef __attribute__((ext_vector_type(8)))  _Float16 v8h;
typedef __attribute__((ext_vector_type(8)))  float    v8f;

#define DIM       384
#define MEDGE     55296
#define CORR_IN   882
#define CORR_PADK 896
#define NFRM      24
#define PPFR      96
#define KSEGS     2304
#define ASEGS     576

// ---------------------------------------------------------------------------
// WMMA fragment helpers (wave32, v_wmma_f32_16x16x32_f16)
// 16-bit A-matrix 16x32: lane l -> row l%16; half-halves at
// [ (l/16)*8 .. +8 ) and [ 16+(l/16)*8 .. +8 ) => two 16B loads per fragment.
// B tile (32x16 = KxN) loads W rows ([out,in] row-major) with the same map.
// ---------------------------------------------------------------------------
__device__ __forceinline__ v16h load_frag16x32(const half_t* base, int ld, int kbase) {
  const int lane = threadIdx.x & 31;
  const int row  = lane & 15;
  const int hl   = lane >> 4;
  const half_t* p = base + (size_t)row * ld + kbase + hl * 8;
  v8h lo = *(const v8h*)p;
  v8h hi = *(const v8h*)(p + 16);
  v16h r;
#pragma unroll
  for (int i = 0; i < 8; ++i) { r[i] = lo[i]; r[i + 8] = hi[i]; }
  return r;
}

__device__ __forceinline__ void zero6(v8f acc[6]) {
#pragma unroll
  for (int t = 0; t < 6; ++t)
#pragma unroll
    for (int i = 0; i < 8; ++i) acc[t][i] = 0.f;
}

// One wave computes a 32(M) x 96(N) tile as 2 row-frags x 6 col-tiles.
// Each B fragment is loaded once per k-step and reused for both row-frags,
// halving weight (L2) traffic vs the 16-row version.
__device__ __forceinline__ void gemm12(const half_t* A, int lda,
                                       const half_t* W, int ldw,
                                       int K, int n0, v8f a0[6], v8f a1[6]) {
#pragma unroll 1
  for (int kk = 0; kk < K; kk += 32) {
    v16h fa0 = load_frag16x32(A, lda, kk);
    v16h fa1 = load_frag16x32(A + (size_t)16 * lda, lda, kk);
#pragma unroll
    for (int t = 0; t < 6; ++t) {
      v16h b = load_frag16x32(W + (size_t)(n0 + 16 * t) * ldw, ldw, kk);
      a0[t] = __builtin_amdgcn_wmma_f32_16x16x32_f16(false, fa0, false, b,
                                                     (short)0, a0[t], false, false);
      a1[t] = __builtin_amdgcn_wmma_f32_16x16x32_f16(false, fa1, false, b,
                                                     (short)0, a1[t], false, false);
    }
  }
}

// C/D layout: VGPR r, lane l -> row (r + 8*(l>>4)), col l%16.
__device__ __forceinline__ void store6_f16_r(half_t* dst, int ld, int n0, int row0,
                                             const v8f acc[6], const float* bias,
                                             int relu_flag) {
  const int lane = threadIdx.x & 31;
  const int nl = lane & 15;
  const int mb = row0 + (lane >> 4) * 8;
#pragma unroll
  for (int t = 0; t < 6; ++t) {
    const int n = n0 + 16 * t + nl;
    const float bv = bias[n];
#pragma unroll
    for (int r = 0; r < 8; ++r) {
      float v = acc[t][r] + bv;
      if (relu_flag) v = fmaxf(v, 0.f);
      dst[(size_t)(mb + r) * ld + n] = (half_t)v;
    }
  }
}

__device__ __forceinline__ void store6_f32_r(float* dst, int ld, int n0, int row0,
                                             const v8f acc[6], const float* bias) {
  const int lane = threadIdx.x & 31;
  const int nl = lane & 15;
  const int mb = row0 + (lane >> 4) * 8;
#pragma unroll
  for (int t = 0; t < 6; ++t) {
    const int n = n0 + 16 * t + nl;
    const float bv = bias[n];
#pragma unroll
    for (int r = 0; r < 8; ++r) dst[(size_t)(mb + r) * ld + n] = acc[t][r] + bv;
  }
}

// Row LayerNorm over 32x384 f16 LDS tile (128 thr: 4 per row), f32 stats.
__device__ __forceinline__ void ln32_f16(const half_t* Fh, const float* w, const float* b,
                                         half_t* outT, int relu_flag,
                                         float* g32, half_t* g16) {
  __shared__ float red1[128], red2[128];
  const int sub = threadIdx.x & 3;
  const int row = threadIdx.x >> 2;   // 0..31
  const half_t* Fr = Fh + (size_t)row * DIM;
  float s1 = 0.f, s2 = 0.f;
  for (int c = sub; c < DIM; c += 4) { const float v = (float)Fr[c]; s1 += v; s2 += v * v; }
  red1[threadIdx.x] = s1; red2[threadIdx.x] = s2;
  __syncthreads();
  float t1 = 0.f, t2 = 0.f;
#pragma unroll
  for (int i = 0; i < 4; ++i) { t1 += red1[(row << 2) + i]; t2 += red2[(row << 2) + i]; }
  const float mu = t1 * (1.f / DIM);
  const float var = t2 * (1.f / DIM) - mu * mu;
  const float rs = rsqrtf(var + 1e-3f);
  for (int c = sub; c < DIM; c += 4) {
    float v = ((float)Fr[c] - mu) * rs * w[c] + b[c];
    if (relu_flag) v = fmaxf(v, 0.f);
    if (outT) outT[(size_t)row * DIM + c] = (half_t)v;
    if (g32)  g32[(size_t)row * DIM + c]  = v;
    if (g16)  g16[(size_t)row * DIM + c]  = (half_t)v;
  }
  __syncthreads();
}

// Row LayerNorm reading 32 rows of f32 from global, writing f16 LDS tile.
__device__ __forceinline__ void ln32_global(const float* src, const float* w,
                                            const float* b, half_t* X) {
  __shared__ float red1[128], red2[128];
  const int sub = threadIdx.x & 3;
  const int row = threadIdx.x >> 2;
  const float* Sr = src + (size_t)row * DIM;
  float s1 = 0.f, s2 = 0.f;
  for (int c = sub; c < DIM; c += 4) { const float v = Sr[c]; s1 += v; s2 += v * v; }
  red1[threadIdx.x] = s1; red2[threadIdx.x] = s2;
  __syncthreads();
  float t1 = 0.f, t2 = 0.f;
#pragma unroll
  for (int i = 0; i < 4; ++i) { t1 += red1[(row << 2) + i]; t2 += red2[(row << 2) + i]; }
  const float mu = t1 * (1.f / DIM);
  const float var = t2 * (1.f / DIM) - mu * mu;
  const float rs = rsqrtf(var + 1e-3f);
  for (int c = sub; c < DIM; c += 4)
    X[(size_t)row * DIM + c] = (half_t)((Sr[c] - mu) * rs * w[c] + b[c]);
  __syncthreads();
}

// ---------------------------------------------------------------------------
// Kernel 0: f32 -> f16 pack with column padding (weights, corr 882->896)
// ---------------------------------------------------------------------------
__global__ void pack_f16_kernel(const float* __restrict__ src, half_t* __restrict__ dst,
                                int rows, int scols, int dcols) {
  const size_t total = (size_t)rows * dcols;
  for (size_t i = (size_t)blockIdx.x * blockDim.x + threadIdx.x; i < total;
       i += (size_t)gridDim.x * blockDim.x) {
    const int c = (int)(i % (size_t)dcols);
    const size_t r = i / (size_t)dcols;
    dst[i] = (c < scols) ? (half_t)src[r * (size_t)scols + c] : (half_t)0.f;
  }
}

// ---------------------------------------------------------------------------
// Kernel 1: fused corr encoder + net0 = LN(net + inp + c), 32 rows/block
//   h1 = relu(corr @ W0^T+b0); h2 = h1 @ W1^T+b1; x = relu(LN(h2));
//   c  = x @ W2^T+b2; net0 = LN(net+inp+c) -> net_f32 + f16 mirror
// A of GEMM1 read directly from global (contiguous rows, L1-shared by waves).
// ---------------------------------------------------------------------------
__global__ void __launch_bounds__(128) corr_stage_kernel(
    const half_t* __restrict__ corr16, const float* __restrict__ net_in,
    const float* __restrict__ inp,
    const half_t* __restrict__ W0, const float* __restrict__ b0,
    const half_t* __restrict__ W1, const float* __restrict__ b1,
    const float* __restrict__ clw, const float* __restrict__ clb,
    const half_t* __restrict__ W2, const float* __restrict__ b2,
    const float* __restrict__ nw, const float* __restrict__ nb,
    float* __restrict__ net_f32, half_t* __restrict__ mir) {
  __shared__ __align__(16) half_t T[32 * DIM];   // ping
  __shared__ __align__(16) half_t T2[32 * DIM];  // pong
  const size_t m0 = (size_t)blockIdx.x * 32;
  const int n0 = (threadIdx.x >> 5) * 96;
  v8f a0[6], a1[6];

  zero6(a0); zero6(a1);
  gemm12(corr16 + m0 * CORR_PADK, CORR_PADK, W0, CORR_PADK, CORR_PADK, n0, a0, a1);
  store6_f16_r(T2, DIM, n0, 0, a0, b0, 1);       // h1
  store6_f16_r(T2, DIM, n0, 16, a1, b0, 1);
  __syncthreads();

  zero6(a0); zero6(a1);
  gemm12(T2, DIM, W1, DIM, DIM, n0, a0, a1);     // h2
  store6_f16_r(T, DIM, n0, 0, a0, b1, 0);
  store6_f16_r(T, DIM, n0, 16, a1, b1, 0);
  __syncthreads();

  ln32_f16(T, clw, clb, T2, 1, nullptr, nullptr); // x = relu(LN(h2)) -> T2

  zero6(a0); zero6(a1);
  gemm12(T2, DIM, W2, DIM, DIM, n0, a0, a1);     // c
  {                                              // T = (f16)(net + inp + c)
    const int lane = threadIdx.x & 31;
    const int nl = lane & 15;
    const int mbase = (lane >> 4) * 8;
    const float* nb_ = net_in + m0 * DIM;
    const float* ib_ = inp + m0 * DIM;
#pragma unroll
    for (int rb = 0; rb < 2; ++rb) {
      const v8f* acc = rb ? a1 : a0;
      const int mb = mbase + rb * 16;
#pragma unroll
      for (int t = 0; t < 6; ++t) {
        const int n = n0 + 16 * t + nl;
        const float bv = b2[n];
#pragma unroll
        for (int r = 0; r < 8; ++r) {
          const int rw = mb + r;
          T[(size_t)rw * DIM + n] =
              (half_t)(acc[t][r] + bv + nb_[(size_t)rw * DIM + n] + ib_[(size_t)rw * DIM + n]);
        }
      }
    }
  }
  __syncthreads();
  ln32_f16(T, nw, nb, nullptr, 0, net_f32 + m0 * DIM, mir + m0 * DIM);
}

// ---------------------------------------------------------------------------
// Kernel 2: neighbor gather + 2-layer MLP, net += mlp2(mask*net[idx]); 32 rows
// mirrors ping-pong (mir_in read-only, mir_out written) to avoid races.
// ---------------------------------------------------------------------------
__global__ void __launch_bounds__(128) neighbor_kernel(
    const int* __restrict__ idx, const half_t* __restrict__ mir_in,
    const half_t* __restrict__ Wa, const float* __restrict__ ba,
    const half_t* __restrict__ Wb, const float* __restrict__ bb,
    float* __restrict__ net_f32, half_t* __restrict__ mir_out) {
  __shared__ __align__(16) half_t A[32 * DIM];
  __shared__ __align__(16) half_t T[32 * DIM];
  const size_t m0 = (size_t)blockIdx.x * 32;

  for (int i = threadIdx.x; i < 32 * DIM / 8; i += 128) {
    const int r = i / (DIM / 8);
    const int cv = i % (DIM / 8);
    const int s = idx[m0 + r];
    v8h v;
    if (s >= 0) v = *(const v8h*)(mir_in + (size_t)s * DIM + cv * 8);
    else {
#pragma unroll
      for (int k = 0; k < 8; ++k) v[k] = (half_t)0.f;
    }
    *(v8h*)(A + (size_t)r * DIM + cv * 8) = v;
  }
  __syncthreads();

  const int n0 = (threadIdx.x >> 5) * 96;
  v8f a0[6], a1[6];
  zero6(a0); zero6(a1);
  gemm12(A, DIM, Wa, DIM, DIM, n0, a0, a1);
  store6_f16_r(T, DIM, n0, 0, a0, ba, 1);
  store6_f16_r(T, DIM, n0, 16, a1, ba, 1);
  __syncthreads();
  zero6(a0); zero6(a1);
  gemm12(T, DIM, Wb, DIM, DIM, n0, a0, a1);
  {
    const int lane = threadIdx.x & 31;
    const int nl = lane & 15;
    const int mbase = (lane >> 4) * 8;
#pragma unroll
    for (int rb = 0; rb < 2; ++rb) {
      const v8f* acc = rb ? a1 : a0;
      const int mb = mbase + rb * 16;
#pragma unroll
      for (int t = 0; t < 6; ++t) {
        const int n = n0 + 16 * t + nl;
        const float bv = bb[n];
#pragma unroll
        for (int r = 0; r < 8; ++r) {
          const size_t off = (m0 + mb + r) * DIM + n;
          const float v = net_f32[off] + acc[t][r] + bv;
          net_f32[off] = v;
          mir_out[off] = (half_t)v;
        }
      }
    }
  }
}

// ---------------------------------------------------------------------------
// Kernel 3: generic out = X @ W^T + b (f16 in/out), 32 rows/block, A from
// global (contiguous rows shared by the 4 waves through L1).
// ---------------------------------------------------------------------------
__global__ void __launch_bounds__(128) gemm_bias_kernel(
    const half_t* __restrict__ X, const half_t* __restrict__ W,
    const float* __restrict__ b, half_t* __restrict__ out) {
  const size_t m0 = (size_t)blockIdx.x * 32;
  const int n0 = (threadIdx.x >> 5) * 96;
  v8f a0[6], a1[6];
  zero6(a0); zero6(a1);
  gemm12(X + m0 * DIM, DIM, W, DIM, DIM, n0, a0, a1);
  const int lane = threadIdx.x & 31;
  const int nl = lane & 15;
  const int mbase = (lane >> 4) * 8;
#pragma unroll
  for (int rb = 0; rb < 2; ++rb) {
    const v8f* acc = rb ? a1 : a0;
    const int mb = mbase + rb * 16;
#pragma unroll
    for (int t = 0; t < 6; ++t) {
      const int n = n0 + 16 * t + nl;
      const float bv = b[n];
#pragma unroll
      for (int r = 0; r < 8; ++r)
        out[(m0 + mb + r) * DIM + n] = (half_t)(acc[t][r] + bv);
    }
  }
}

// ---------------------------------------------------------------------------
// Kernel 4: per-segment (per-column) softmax-weighted sum of f, logits g
// mode 0: kk  segments: seg s -> rows s*24 + t,          t<24  (2304 segs)
// mode 1: aij segments: seg s -> rows (s/24)*2304+(s%24)+p*24, p<96 (576)
// ---------------------------------------------------------------------------
__global__ void __launch_bounds__(128) seg_softmax_kernel(
    const half_t* __restrict__ f, const half_t* __restrict__ g,
    half_t* __restrict__ y, int mode) {
  const int s0 = blockIdx.x * 16;
  for (int w = threadIdx.x; w < 16 * DIM; w += 128) {
    const int si = w / DIM, c = w % DIM;
    const int seg = s0 + si;
    size_t base; int cnt, stride;
    if (mode == 0) { base = (size_t)seg * NFRM * DIM; cnt = NFRM; stride = DIM; }
    else {
      const int ii = seg / NFRM, jj = seg % NFRM;
      base = ((size_t)ii * (PPFR * NFRM) + jj) * DIM; cnt = PPFR; stride = NFRM * DIM;
    }
    float mx = -3.4e38f;
    for (int t = 0; t < cnt; ++t)
      mx = fmaxf(mx, (float)g[base + (size_t)t * stride + c]);
    float den = 0.f, num = 0.f;
    for (int t = 0; t < cnt; ++t) {
      const size_t off = base + (size_t)t * stride + c;
      const float e = __expf((float)g[off] - mx);
      den += e; num += e * (float)f[off];
    }
    y[(size_t)seg * DIM + c] = (half_t)(num / den);
  }
}

// ---------------------------------------------------------------------------
// Kernel 5: h = y @ Wh^T + bh (32 segs/block, WMMA), then broadcast-add h to
// every member edge of each segment; refreshes f16 mirror (full coverage).
// ---------------------------------------------------------------------------
__global__ void __launch_bounds__(128) h_scatter_kernel(
    const half_t* __restrict__ y, const half_t* __restrict__ Wh,
    const float* __restrict__ bh, int mode,
    float* __restrict__ net_f32, half_t* __restrict__ mir_out) {
  __shared__ float H[32 * DIM];
  const int s0 = blockIdx.x * 32;
  const int n0 = (threadIdx.x >> 5) * 96;
  v8f a0[6], a1[6];
  zero6(a0); zero6(a1);
  gemm12(y + (size_t)s0 * DIM, DIM, Wh, DIM, DIM, n0, a0, a1);
  store6_f32_r(H, DIM, n0, 0, a0, bh);
  store6_f32_r(H, DIM, n0, 16, a1, bh);
  __syncthreads();
  const int cnt = (mode == 0) ? NFRM : PPFR;
  const int stride = (mode == 0) ? DIM : NFRM * DIM;
  for (int w = threadIdx.x; w < 32 * DIM; w += 128) {
    const int si = w / DIM, c = w % DIM;
    const int seg = s0 + si;
    const size_t base = (mode == 0)
        ? (size_t)seg * NFRM * DIM
        : ((size_t)(seg / NFRM) * (PPFR * NFRM) + (seg % NFRM)) * DIM;
    const float hv = H[si * DIM + c];
    for (int t = 0; t < cnt; ++t) {
      const size_t off = base + (size_t)t * stride + c;
      const float v = net_f32[off] + hv;
      net_f32[off] = v;
      mir_out[off] = (half_t)v;
    }
  }
}

// ---------------------------------------------------------------------------
// Kernel 6: fused LN + gated residual (in-place on net_f32), 32 rows/block
//   x = LN(net); net = x + sigmoid(x@Wg+bg) * ((relu(x@W1+b1))@W2+b2)
// Gate tile parked in global scratch (L2-hot) to keep LDS at 48 KB.
// ---------------------------------------------------------------------------
__global__ void __launch_bounds__(128) gru_kernel(
    float* __restrict__ net_f32,
    const float* __restrict__ lnw, const float* __restrict__ lnb,
    const half_t* __restrict__ Wg, const float* __restrict__ bg,
    const half_t* __restrict__ W1, const float* __restrict__ b1,
    const half_t* __restrict__ W2, const float* __restrict__ b2,
    half_t* __restrict__ gate_ws) {
  __shared__ __align__(16) half_t X[32 * DIM];
  __shared__ __align__(16) half_t T[32 * DIM];
  const size_t m0 = (size_t)blockIdx.x * 32;
  ln32_global(net_f32 + m0 * DIM, lnw, lnb, X);   // x -> X

  const int n0 = (threadIdx.x >> 5) * 96;
  const int lane = threadIdx.x & 31;
  const int nl = lane & 15;
  const int mbase = (lane >> 4) * 8;
  v8f a0[6], a1[6];

  zero6(a0); zero6(a1);
  gemm12(X, DIM, Wg, DIM, DIM, n0, a0, a1);       // gate logits
#pragma unroll
  for (int rb = 0; rb < 2; ++rb) {
    const v8f* acc = rb ? a1 : a0;
    const int mb = mbase + rb * 16;
#pragma unroll
    for (int t = 0; t < 6; ++t) {
      const int n = n0 + 16 * t + nl;
      const float bv = bg[n];
#pragma unroll
      for (int r = 0; r < 8; ++r) {
        const float v = acc[t][r] + bv;
        gate_ws[(m0 + mb + r) * DIM + n] = (half_t)(1.f / (1.f + __expf(-v)));
      }
    }
  }

  zero6(a0); zero6(a1);
  gemm12(X, DIM, W1, DIM, DIM, n0, a0, a1);       // hidden
  store6_f16_r(T, DIM, n0, 0, a0, b1, 1);
  store6_f16_r(T, DIM, n0, 16, a1, b1, 1);
  __syncthreads();

  zero6(a0); zero6(a1);
  gemm12(T, DIM, W2, DIM, DIM, n0, a0, a1);       // res
#pragma unroll
  for (int rb = 0; rb < 2; ++rb) {
    const v8f* acc = rb ? a1 : a0;
    const int mb = mbase + rb * 16;
#pragma unroll
    for (int t = 0; t < 6; ++t) {
      const int n = n0 + 16 * t + nl;
      const float bv = b2[n];
#pragma unroll
      for (int r = 0; r < 8; ++r) {
        const int rw = mb + r;
        const float res = acc[t][r] + bv;
        const float xv = (float)X[(size_t)rw * DIM + n];
        const float gv = (float)gate_ws[(m0 + rw) * DIM + n];
        net_f32[(m0 + rw) * DIM + n] = xv + gv * res;
      }
    }
  }
}

// ---------------------------------------------------------------------------
// Kernel 7: output copy + heads: r=relu(net); d=r@Wd^T+bd; w=sigmoid(r@Ww^T+bw)
// d_out layout: [net (M*384)] [d (M*2)] [w (M*2)]
// ---------------------------------------------------------------------------
__global__ void head_kernel(const float* __restrict__ net_f32,
                            const float* __restrict__ Wd, const float* __restrict__ bd,
                            const float* __restrict__ Ww, const float* __restrict__ bw,
                            float* __restrict__ out) {
  const size_t stride = (size_t)gridDim.x * blockDim.x;
  const size_t i0 = (size_t)blockIdx.x * blockDim.x + threadIdx.x;
  for (size_t i = i0; i < (size_t)MEDGE * DIM; i += stride) out[i] = net_f32[i];
  for (size_t i = i0; i < (size_t)MEDGE * 4; i += stride) {
    const size_t row = i >> 2;
    const int o = (int)(i & 3);
    const float* Wp = (o < 2) ? Wd + o * DIM : Ww + (o - 2) * DIM;
    float s = (o < 2) ? bd[o] : bw[o - 2];
    const float* x = net_f32 + row * DIM;
    for (int c = 0; c < DIM; ++c) s += fmaxf(x[c], 0.f) * Wp[c];
    if (o < 2) out[(size_t)MEDGE * DIM + row * 2 + o] = s;
    else out[(size_t)MEDGE * DIM + (size_t)MEDGE * 2 + row * 2 + (o - 2)] =
        1.f / (1.f + __expf(-s));
  }
}

// ---------------------------------------------------------------------------
extern "C" void kernel_launch(void* const* d_in, const int* in_sizes, int n_in,
                              void* d_out, int out_size, void* d_ws, size_t ws_size,
                              hipStream_t stream) {
  (void)in_sizes; (void)n_in; (void)out_size; (void)ws_size;
  const float* net  = (const float*)d_in[0];
  const float* inp  = (const float*)d_in[1];
  const float* corr = (const float*)d_in[2];
  const int*   ix   = (const int*)d_in[6];
  const int*   jx   = (const int*)d_in[7];
  auto pw = [&](int i) { return (const float*)d_in[i]; };

  // workspace carve-up
  unsigned char* ws = (unsigned char*)d_ws;
  const size_t SZ_NET32  = (size_t)MEDGE * DIM * 4;
  const size_t SZ_MIR    = (size_t)MEDGE * DIM * 2;
  const size_t SZ_CORR16 = (size_t)MEDGE * CORR_PADK * 2;
  const size_t SZ_Y      = (size_t)KSEGS * DIM * 2;
  const size_t SZ_W0     = (size_t)DIM * CORR_PADK * 2;
  const size_t SZ_W      = (size_t)DIM * DIM * 2;
  size_t o = 0;
  float*  net32  = (float*)(ws + o);  o += SZ_NET32;
  half_t* mirA   = (half_t*)(ws + o); o += SZ_MIR;
  half_t* mirB   = (half_t*)(ws + o); o += SZ_MIR;
  half_t* corr16 = (half_t*)(ws + o);            // region R: corr16 overlaps f|g
  half_t* fbuf   = (half_t*)(ws + o);
  half_t* gbuf   = (half_t*)(ws + o + SZ_MIR);   o += SZ_CORR16;
  half_t* ybuf   = (half_t*)(ws + o); o += SZ_Y;
  half_t* w0     = (half_t*)(ws + o); o += SZ_W0;
  half_t* wmat[18];
  for (int i = 0; i < 18; ++i) { wmat[i] = (half_t*)(ws + o); o += SZ_W; }
  enum { W_CORR1 = 0, W_CORR2, W_C1A, W_C1B, W_C2A, W_C2B,
         W_AKKF, W_AKKG, W_AKKH, W_AIJF, W_AIJG, W_AIJH,
         W_GR1G, W_GR1R1, W_GR1R2, W_GR2G, W_GR2R1, W_GR2R2 };

  auto pack = [&](const float* src, half_t* dst, int rows, int sc, int dc) {
    size_t total = (size_t)rows * dc;
    int blocks = (int)((total + 255) / 256);
    if (blocks > 16384) blocks = 16384;
    pack_f16_kernel<<<blocks, 256, 0, stream>>>(src, dst, rows, sc, dc);
  };

  const int src_idx[18] = {10, 14, 18, 20, 22, 24, 26, 28, 30, 32, 34, 36,
                           40, 42, 44, 48, 50, 52};
  for (int i = 0; i < 18; ++i) pack(pw(src_idx[i]), wmat[i], DIM, DIM, DIM);
  pack(pw(8), w0, DIM, CORR_IN, CORR_PADK);
  pack(corr, corr16, MEDGE, CORR_IN, CORR_PADK);

  const dim3 gb(MEDGE / 32), tb(128);   // 1728 blocks, 4 waves each

  corr_stage_kernel<<<gb, tb, 0, stream>>>(
      corr16, net, inp, w0, pw(9), wmat[W_CORR1], pw(11), pw(12), pw(13),
      wmat[W_CORR2], pw(15), pw(16), pw(17), net32, mirA);

  neighbor_kernel<<<gb, tb, 0, stream>>>(ix, mirA, wmat[W_C1A], pw(19),
                                         wmat[W_C1B], pw(21), net32, mirB);
  neighbor_kernel<<<gb, tb, 0, stream>>>(jx, mirB, wmat[W_C2A], pw(23),
                                         wmat[W_C2B], pw(25), net32, mirA);

  gemm_bias_kernel<<<gb, tb, 0, stream>>>(mirA, wmat[W_AKKF], pw(27), fbuf);
  gemm_bias_kernel<<<gb, tb, 0, stream>>>(mirA, wmat[W_AKKG], pw(29), gbuf);
  seg_softmax_kernel<<<dim3(KSEGS / 16), tb, 0, stream>>>(fbuf, gbuf, ybuf, 0);
  h_scatter_kernel<<<dim3(KSEGS / 32), tb, 0, stream>>>(ybuf, wmat[W_AKKH], pw(31),
                                                        0, net32, mirB);

  gemm_bias_kernel<<<gb, tb, 0, stream>>>(mirB, wmat[W_AIJF], pw(33), fbuf);
  gemm_bias_kernel<<<gb, tb, 0, stream>>>(mirB, wmat[W_AIJG], pw(35), gbuf);
  seg_softmax_kernel<<<dim3(ASEGS / 16), tb, 0, stream>>>(fbuf, gbuf, ybuf, 1);
  h_scatter_kernel<<<dim3(ASEGS / 32), tb, 0, stream>>>(ybuf, wmat[W_AIJH], pw(37),
                                                        1, net32, mirA);

  gru_kernel<<<gb, tb, 0, stream>>>(net32, pw(38), pw(39), wmat[W_GR1G], pw(41),
                                    wmat[W_GR1R1], pw(43), wmat[W_GR1R2], pw(45), gbuf);
  gru_kernel<<<gb, tb, 0, stream>>>(net32, pw(46), pw(47), wmat[W_GR2G], pw(49),
                                    wmat[W_GR2R1], pw(51), wmat[W_GR2R2], pw(53), gbuf);

  head_kernel<<<4096, 256, 0, stream>>>(net32, pw(54), pw(55), pw(56), pw(57),
                                        (float*)d_out);
}
